// SemanticPromptGenerator_62654982914316
// MI455X (gfx1250) — matrix operand order repaired and checked by reading, not compiled, not run on hardware
//
#include <hip/hip_runtime.h>

#define B_ 2048
#define D_ 768
#define C_ 1024
#define E_ 32768

typedef __attribute__((ext_vector_type(16))) __bf16 v16bf;
typedef __attribute__((ext_vector_type(8)))  float  v8f;

union Frag { uint4 q[2]; v16bf b; };

__device__ __forceinline__ unsigned short f2bf(float f) {
    unsigned u = __float_as_uint(f);
    unsigned r = (u + 0x7FFFu + ((u >> 16) & 1u)) >> 16;   // RNE
    return (unsigned short)r;
}
// monotone float<->uint encoding so unsigned atomicMax == float max (exact, deterministic)
__device__ __forceinline__ unsigned encf(float f) {
    unsigned b = __float_as_uint(f);
    return (b & 0x80000000u) ? ~b : (b | 0x80000000u);
}
__device__ __forceinline__ float decf(unsigned u) {
    unsigned b = (u & 0x80000000u) ? (u ^ 0x80000000u) : ~u;
    return __uint_as_float(b);
}

// CDNA5 async memory->LDS (ASYNCcnt-tracked, bypasses VGPRs) — see CDNA5 ISA
__device__ __forceinline__ void async_ld_b128(unsigned lds_off, unsigned long long gaddr) {
    asm volatile("global_load_async_to_lds_b128 %0, %1, off"
                 :: "v"(lds_off), "v"(gaddr) : "memory");
}
__device__ __forceinline__ void wait_async0() {
    asm volatile("s_wait_asynccnt 0" ::: "memory");
}

// A fragment (16x32 bf16, row-major source, ld = K): lane<16 -> K 0..15, lane>=16 -> K 16..31
__device__ __forceinline__ v16bf load_fragA(const unsigned short* base, int ld, int row, int k, int half) {
    const uint4* p = (const uint4*)(base + (size_t)row * ld + k + half * 16);
    Frag f; f.q[0] = p[0]; f.q[1] = p[1];
    return f.b;
}
// B fragment from B^T row-major (row = output col n, ld = K): lane<16 -> K 0..15, lane>=16 -> K 16..31
__device__ __forceinline__ v16bf load_fragB(const unsigned short* base, int ld, int col, int k, int half) {
    const uint4* p = (const uint4*)(base + (size_t)col * ld + k + half * 16);
    Frag f; f.q[0] = p[0]; f.q[1] = p[1];
    return f.b;
}

// ---------------- prep kernels ----------------

__global__ void k_query(const float* __restrict__ text, const float* __restrict__ img,
                        unsigned short* __restrict__ qbf) {
    int b = blockIdx.x, tid = threadIdx.x;
    __shared__ float red[256];
    float v[3]; float loc = 0.f;
    #pragma unroll
    for (int j = 0; j < 3; ++j) {
        int i = tid + j * 256;
        v[j] = text[(size_t)b * D_ + i] + img[(size_t)b * D_ + i];
        loc += v[j] * v[j];
    }
    red[tid] = loc; __syncthreads();
    for (int s = 128; s > 0; s >>= 1) { if (tid < s) red[tid] += red[tid + s]; __syncthreads(); }
    float rn = rsqrtf(red[0]);
    #pragma unroll
    for (int j = 0; j < 3; ++j) { int i = tid + j * 256; qbf[(size_t)b * D_ + i] = f2bf(v[j] * rn); }
}

__global__ void k_init(unsigned* __restrict__ enc, int* __restrict__ counts) {
    int idx = blockIdx.x * 256 + threadIdx.x;
    if (idx < B_ * C_) enc[idx] = 0u;          // encodes "smaller than any float"
    if (idx < C_) counts[idx] = 0;
}

__global__ void k_cvt4(const float4* __restrict__ src, ushort4* __restrict__ dst, int n4) {
    int i = blockIdx.x * 256 + threadIdx.x;
    if (i < n4) {
        float4 v = src[i];
        ushort4 o; o.x = f2bf(v.x); o.y = f2bf(v.y); o.z = f2bf(v.z); o.w = f2bf(v.w);
        dst[i] = o;
    }
}

__global__ void k_cvt_cn(const float* __restrict__ cn, unsigned short* __restrict__ cnb,
                         unsigned short* __restrict__ cnT) {
    int i = blockIdx.x * 256 + threadIdx.x;
    if (i < C_ * D_) {
        int c = i / D_, d = i - c * D_;
        unsigned short v = f2bf(cn[i]);
        cnb[i] = v;
        cnT[(size_t)d * C_ + c] = v;
    }
}

__global__ void k_count(const int* __restrict__ seg, int* __restrict__ counts) {
    int e = blockIdx.x * 256 + threadIdx.x;
    if (e < E_) atomicAdd(&counts[seg[e]], 1);
}

// ---------------- generic bf16 WMMA GEMM: Out[M,N] = A[M,K] * Bt[N,K]^T ----------------
// one wave per 16x16 tile; ldA = ldBt = K
__global__ void k_gemm_bf16(const unsigned short* __restrict__ A, const unsigned short* __restrict__ Bt,
                            float* __restrict__ Out, int N, int K, int ldo) {
    int wave = (blockIdx.x * blockDim.x + threadIdx.x) >> 5;
    int lane = threadIdx.x & 31;
    int lrow = lane & 15, half = lane >> 4;
    int ntiles = N >> 4;
    int m0 = (wave / ntiles) << 4;
    int n0 = (wave % ntiles) << 4;
    v8f acc;
    #pragma unroll
    for (int i = 0; i < 8; ++i) acc[i] = 0.f;
    for (int k = 0; k < K; k += 32) {
        v16bf a = load_fragA(A,  K, m0 + lrow, k, half);
        v16bf b = load_fragB(Bt, K, n0 + lrow, k, half);
        acc = __builtin_amdgcn_wmma_f32_16x16x32_bf16(false, a, false, b, (short)0, acc, false, false);
    }
    #pragma unroll
    for (int i = 0; i < 8; ++i) {
        int row = m0 + i + half * 8;
        Out[(size_t)row * ldo + n0 + lrow] = acc[i];
    }
}

// ---------------- fused big GEMM + segmented max ----------------
// grid: (E_/256, B_/128), 256 threads (8 waves). Wave w owns rows m0..m0+15, all 256 cols of stripe.
// ext tile double-buffered in LDS via async-to-LDS; XOR swizzle kb ^ ((r>>4)&7) ^ (r&7) keeps both
// the write pattern and the strided B-fragment reads bank-friendly.
// Column interleave: tile t, lane lrow -> column e0 + 16*lrow + t  (contiguous per lane across t).
__global__ void k_simmax(const unsigned short* __restrict__ qbf,
                         const unsigned short* __restrict__ extbf,
                         const int* __restrict__ seg,
                         unsigned* __restrict__ enc) {
    __shared__ uint4 tile[2][256 * 8];    // 2 x (256 ext rows x 64 bf16) = 64KB
    int tid  = threadIdx.x;
    int lane = tid & 31, wave = tid >> 5;
    int lrow = lane & 15, half = lane >> 4;
    int e0 = blockIdx.x * 256;
    int m0 = blockIdx.y * 128 + wave * 16;

    v8f acc[16];
    #pragma unroll
    for (int t = 0; t < 16; ++t)
        #pragma unroll
        for (int i = 0; i < 8; ++i) acc[t][i] = 0.f;

    int swz_st = ((tid >> 4) & 7) ^ (tid & 7);
    const unsigned short* rowp = extbf + (size_t)(e0 + tid) * D_;
    unsigned ldsrow0 = (unsigned)(uintptr_t)(&tile[0][tid * 8]);
    unsigned ldsrow1 = (unsigned)(uintptr_t)(&tile[1][tid * 8]);

    // prefetch chunk 0 into buffer 0
    #pragma unroll
    for (int kb = 0; kb < 8; ++kb)
        async_ld_b128(ldsrow0 + (unsigned)((kb ^ swz_st) << 4),
                      (unsigned long long)(uintptr_t)(rowp + kb * 8));

    int buf = 0;
    for (int kc = 0; kc < D_; kc += 64) {
        wait_async0();          // my rows for the current buffer have landed in LDS
        __syncthreads();        // everyone's rows have landed; prior compute on spare buf done
        if (kc + 64 < D_) {
            unsigned dst = buf ? ldsrow0 : ldsrow1;
            #pragma unroll
            for (int kb = 0; kb < 8; ++kb)
                async_ld_b128(dst + (unsigned)((kb ^ swz_st) << 4),
                              (unsigned long long)(uintptr_t)(rowp + kc + 64 + kb * 8));
        }
        const uint4* tb = tile[buf];
        #pragma unroll
        for (int ks = 0; ks < 2; ++ks) {
            v16bf a = load_fragA(qbf, D_, m0 + lrow, kc + ks * 32, half);
            int kb0 = ks * 4 + half * 2;
            #pragma unroll
            for (int t = 0; t < 16; ++t) {
                int r  = (lrow << 4) + t;                // ext row within stripe for this lane/tile
                int sw = (lrow & 7) ^ (t & 7);
                Frag f;
                f.q[0] = tb[r * 8 + (kb0 ^ sw)];
                f.q[1] = tb[r * 8 + ((kb0 + 1) ^ sw)];
                acc[t] = __builtin_amdgcn_wmma_f32_16x16x32_bf16(false, a, false, f.b, (short)0,
                                                                 acc[t], false, false);
            }
        }
        buf ^= 1;
    }

    // segmented running max over this lane's 16 contiguous columns, flush via encoded atomicMax
    int colbase = e0 + (lrow << 4);
    float run[8];
    int curseg = seg[colbase];
    #pragma unroll
    for (int i = 0; i < 8; ++i) run[i] = acc[0][i];
    #pragma unroll
    for (int t = 1; t < 16; ++t) {
        int sgt = seg[colbase + t];
        if (sgt == curseg) {
            #pragma unroll
            for (int i = 0; i < 8; ++i) run[i] = fmaxf(run[i], acc[t][i]);
        } else {
            #pragma unroll
            for (int i = 0; i < 8; ++i) {
                int row = m0 + i + half * 8;
                atomicMax(&enc[(size_t)row * C_ + curseg], encf(run[i]));
            }
            curseg = sgt;
            #pragma unroll
            for (int i = 0; i < 8; ++i) run[i] = acc[t][i];
        }
    }
    #pragma unroll
    for (int i = 0; i < 8; ++i) {
        int row = m0 + i + half * 8;
        atomicMax(&enc[(size_t)row * C_ + curseg], encf(run[i]));
    }
}

// ---------------- softmax row kernel ----------------
__global__ void k_softmax(const unsigned* __restrict__ enc, const int* __restrict__ counts,
                          const float* __restrict__ name_sim, const float* __restrict__ temp,
                          unsigned short* __restrict__ wbf) {
    int b = blockIdx.x, tid = threadIdx.x;
    __shared__ float red[256];
    float invT = 1.0f / temp[0];
    float z[4]; float mx = -3.0e38f;
    #pragma unroll
    for (int j = 0; j < 4; ++j) {
        int c = tid + j * 256;
        float sc = (counts[c] > 0) ? decf(enc[(size_t)b * C_ + c]) : name_sim[(size_t)b * C_ + c];
        z[j] = sc * invT;
        mx = fmaxf(mx, z[j]);
    }
    red[tid] = mx; __syncthreads();
    for (int st = 128; st > 0; st >>= 1) { if (tid < st) red[tid] = fmaxf(red[tid], red[tid + st]); __syncthreads(); }
    mx = red[0]; __syncthreads();
    float e[4]; float sum = 0.f;
    #pragma unroll
    for (int j = 0; j < 4; ++j) { e[j] = __expf(z[j] - mx); sum += e[j]; }
    red[tid] = sum; __syncthreads();
    for (int st = 128; st > 0; st >>= 1) { if (tid < st) red[tid] += red[tid + st]; __syncthreads(); }
    float rs = 1.0f / red[0];
    #pragma unroll
    for (int j = 0; j < 4; ++j) { int c = tid + j * 256; wbf[(size_t)b * C_ + c] = f2bf(e[j] * rs); }
}

// ---------------- finalize: out = l2norm(text + dyn) ----------------
__global__ void k_final(const float* __restrict__ text, const float* __restrict__ dyn,
                        float* __restrict__ out) {
    int b = blockIdx.x, tid = threadIdx.x;
    __shared__ float red[256];
    float v[3]; float loc = 0.f;
    #pragma unroll
    for (int j = 0; j < 3; ++j) {
        int i = tid + j * 256;
        v[j] = text[(size_t)b * D_ + i] + dyn[(size_t)b * D_ + i];
        loc += v[j] * v[j];
    }
    red[tid] = loc; __syncthreads();
    for (int st = 128; st > 0; st >>= 1) { if (tid < st) red[tid] += red[tid + st]; __syncthreads(); }
    float rn = rsqrtf(red[0]);
    #pragma unroll
    for (int j = 0; j < 3; ++j) { int i = tid + j * 256; out[(size_t)b * D_ + i] = v[j] * rn; }
}

extern "C" void kernel_launch(void* const* d_in, const int* in_sizes, int n_in,
                              void* d_out, int out_size, void* d_ws, size_t ws_size,
                              hipStream_t stream) {
    const float* text = (const float*)d_in[0];
    const float* img  = (const float*)d_in[1];
    const float* cn   = (const float*)d_in[2];
    const float* ext  = (const float*)d_in[3];
    const float* temp = (const float*)d_in[4];
    const int*   seg  = (const int*)d_in[5];
    float* out = (float*)d_out;

    char* p = (char*)d_ws;
    auto alloc = [&](size_t bytes) -> char* {
        char* r = p; p += (bytes + 255) & ~(size_t)255; return r;
    };
    unsigned short* qbf   = (unsigned short*)alloc((size_t)B_ * D_ * 2);
    unsigned short* extbf = (unsigned short*)alloc((size_t)E_ * D_ * 2);
    unsigned short* cnb   = (unsigned short*)alloc((size_t)C_ * D_ * 2);
    unsigned short* cnT   = (unsigned short*)alloc((size_t)D_ * C_ * 2);
    float*          nsim  = (float*)alloc((size_t)B_ * C_ * 4);
    unsigned*       enc   = (unsigned*)alloc((size_t)B_ * C_ * 4);
    int*            cnts  = (int*)alloc((size_t)C_ * 4);
    unsigned short* wbf   = (unsigned short*)alloc((size_t)B_ * C_ * 2);
    float*          dyn   = (float*)alloc((size_t)B_ * D_ * 4);

    k_query <<<B_, 256, 0, stream>>>(text, img, qbf);
    k_init  <<<(B_ * C_) / 256, 256, 0, stream>>>(enc, cnts);
    k_cvt4  <<<((E_ * D_ / 4) + 255) / 256, 256, 0, stream>>>((const float4*)ext, (ushort4*)extbf, E_ * D_ / 4);
    k_cvt_cn<<<(C_ * D_) / 256, 256, 0, stream>>>(cn, cnb, cnT);
    k_count <<<E_ / 256, 256, 0, stream>>>(seg, cnts);
    // name_sim[B,C] = q @ cn^T   (K = D)
    k_gemm_bf16<<<(B_ / 16) * (C_ / 16) / 8, 256, 0, stream>>>(qbf, cnb, nsim, C_, D_, C_);
    // fused sim GEMM + segmented max  (K = D, 103 GFLOP)
    k_simmax<<<dim3(E_ / 256, B_ / 128), 256, 0, stream>>>(qbf, extbf, seg, enc);
    k_softmax<<<B_, 256, 0, stream>>>(enc, cnts, nsim, temp, wbf);
    // dyn[B,D] = weights @ cn    (K = C, cnT is cn^T so Bt rows are output cols)
    k_gemm_bf16<<<(B_ / 16) * (D_ / 16) / 8, 256, 0, stream>>>(wbf, cnT, dyn, D_, C_, D_);
    k_final<<<B_, 256, 0, stream>>>(text, dyn, out);
}
